// RelativeSelfAttention_16965120820041
// MI455X (gfx1250) — matrix operand order repaired and checked by reading, not compiled, not run on hardware
//
#include <hip/hip_runtime.h>

typedef __attribute__((ext_vector_type(16))) __bf16 v16bf;
typedef __attribute__((ext_vector_type(8)))  float  v8f;

#define NW   49      // tokens per window
#define NP   64      // padded tokens
#define CH   128     // channels
#define QKD  256     // 2*qkdim
#define NH   4       // heads
#define HD   32      // head dim
#define NWIN 4096
#define SCALE 0.17677669529663687f   // 32^-0.5

// LDS strides padded for 64-bank (256B period) conflict avoidance
#define LDA 136      // bf16 row stride for xb/qs/vs   (272 B -> bank step 4)
#define KLD 72       // bf16 token stride for kT       (144 B)
#define NPS 65       // f32 row stride for scores      (260 B, odd dwords)

// ---------------- fragment loaders (ISA 7.12.2 layouts, wave32) ----------------

// A-matrix 16x32 bf16 from row-major [ld] storage at (row0, k0)
__device__ __forceinline__ v16bf load_a_bf16(const __bf16* base, int row0, int ld,
                                             int k0, int lane) {
  v16bf a;
  const int m  = row0 + (lane & 15);
  const int hi = (lane >> 4) ? 8 : 0;
  const __bf16* p = base + m * ld + k0;
#pragma unroll
  for (int v = 0; v < 8; ++v) {
    const int kb = (v < 4) ? (hi + 2 * v) : (16 + hi + 2 * (v - 4));
    a[2 * v]     = p[kb];
    a[2 * v + 1] = p[kb + 1];
  }
  return a;
}

// A-matrix 16x32 built from an f32 row-major matrix (converted in-register)
__device__ __forceinline__ v16bf load_a_f32(const float* base, int row0, int ld,
                                            int k0, int lane) {
  v16bf a;
  const int m  = row0 + (lane & 15);
  const int hi = (lane >> 4) ? 8 : 0;
  const float* p = base + m * ld + k0;
#pragma unroll
  for (int v = 0; v < 8; ++v) {
    const int kb = (v < 4) ? (hi + 2 * v) : (16 + hi + 2 * (v - 4));
    a[2 * v]     = (__bf16)p[kb];
    a[2 * v + 1] = (__bf16)p[kb + 1];
  }
  return a;
}

// B-matrix 32x16 bf16 from row-major [k][ld] storage: lane column n0+(lane&15),
// k in table order (same as A).
__device__ __forceinline__ v16bf load_b_bf16(const __bf16* base, int k0, int ld,
                                             int n0, int lane) {
  v16bf b;
  const int hi = (lane >> 4) ? 8 : 0;
  const __bf16* p = base + (size_t)k0 * ld + n0 + (lane & 15);
#pragma unroll
  for (int s = 0; s < 16; ++s) {
    const int k = (s < 8) ? (hi + s) : (16 + hi + (s - 8));
    b[s] = p[(size_t)k * ld];
  }
  return b;
}

// ---------------- prep: transpose+convert weights, gather bias ----------------

__global__ void prep_kernel(const float* __restrict__ qk_w,   // [256][128]
                            const float* __restrict__ v_w,    // [128][128]
                            const float* __restrict__ proj_w, // [128][128]
                            const float* __restrict__ bias_table, // [169][4]
                            const int*   __restrict__ rel_idx,    // [49*49]
                            __bf16* qk_wt,   // [128][256]  (K-major)
                            __bf16* v_wt,    // [128][128]
                            __bf16* proj_wt, // [128][128]
                            float*  bias_full /* [4][49][49] */) {
  const int i = blockIdx.x * blockDim.x + threadIdx.x;
  if (i < CH * QKD) {
    const int k = i / QKD, n = i % QKD;
    qk_wt[i] = (__bf16)qk_w[n * CH + k];
  }
  if (i < CH * CH) {
    const int k = i / CH, n = i % CH;
    v_wt[i]    = (__bf16)v_w[n * CH + k];
    proj_wt[i] = (__bf16)proj_w[n * CH + k];
  }
  if (i < NH * NW * NW) {
    const int h = i / (NW * NW), r = i % (NW * NW);
    bias_full[i] = bias_table[rel_idx[r] * NH + h];
  }
}

// ---------------- fused windowed attention ----------------

__global__ __launch_bounds__(256)
void attn_fused_kernel(const float* __restrict__ x,        // [B][49][128]
                       const float* __restrict__ qk_b,     // [256]
                       const float* __restrict__ v_b,      // [128]
                       const float* __restrict__ proj_b,   // [128]
                       const __bf16* __restrict__ qk_wt,   // [128][256]
                       const __bf16* __restrict__ v_wt,    // [128][128]
                       const __bf16* __restrict__ proj_wt, // [128][128]
                       const float* __restrict__ bias_full,// [4][49][49]
                       float* __restrict__ out)             // [B][49][128]
{
  __shared__ __align__(16) __bf16 xb[NP * LDA];        // 17 KB: x bf16, later head-concat
  __shared__ __align__(16) __bf16 qs[NP * LDA];        // 17 KB: q scaled, [tok][h*32+d]
  __shared__ __align__(16) __bf16 kts[NH * HD * KLD];  // 18 KB: kT, [h][d][tok]
  __shared__ __align__(16) __bf16 vs[NP * LDA];        // 17 KB: v, zero-padded rows
  __shared__ __align__(16) float  sc[NH * NP * NPS];   // 65 KB: logits/probs; doubles as
                                                       //        f32 staging for async x copy
  const int win  = blockIdx.x;
  const int tid  = threadIdx.x;
  const int lane = tid & 31;
  const int wave = tid >> 5;     // 0..7

  // ---- async-stage x window (f32) into LDS via GLOBAL_LOAD_ASYNC_TO_LDS ----
  float* xstage = sc;                           // dead storage until scores phase
  const float* xg = x + (size_t)win * NW * CH;
  for (int i = tid; i < NW * CH / 4; i += 256) {           // 1568 x b128 transfers
    const unsigned long long gaddr = (unsigned long long)(uintptr_t)(xg + i * 4);
    const unsigned           laddr = (unsigned)(uintptr_t)(xstage + i * 4);
    asm volatile("global_load_async_to_lds_b128 %0, %1, off"
                 :: "v"(laddr), "v"(gaddr) : "memory");
  }
  // zero-pad xb rows 49..63 while the DMA is in flight
  for (int i = tid; i < (NP - NW) * CH; i += 256) {
    const int m = NW + i / CH, c = i % CH;
    xb[m * LDA + c] = (__bf16)0.f;
  }
  asm volatile("s_wait_asynccnt 0x0" ::: "memory");
  __syncthreads();

  // ---- convert staged f32 -> bf16 tile ----
  for (int i = tid; i < NW * CH; i += 256) {
    const int m = i >> 7, c = i & 127;
    xb[m * LDA + c] = (__bf16)xstage[i];
  }
  __syncthreads();

  const int nlo = lane & 15;
  const int mhi = (lane >> 4) * 8;

  // ---- fused QK projection: [64x128] @ [128x256] -> q (scaled) + kT ----
  for (int t = wave; t < 64; t += 8) {
    const int m0 = (t >> 4) * 16;
    const int n0 = (t & 15) * 16;
    v8f acc = {};
#pragma unroll
    for (int kk = 0; kk < CH; kk += 32) {
      v16bf a = load_a_bf16(xb, m0, LDA, kk, lane);
      v16bf b = load_b_bf16(qk_wt, kk, QKD, n0, lane);
      acc = __builtin_amdgcn_wmma_f32_16x16x32_bf16(false, a, false, b,
                                                    (short)0, acc, false, false);
    }
    const int n  = n0 + nlo;
    const float bv = qk_b[n];
    if (n0 < CH) {            // q half: pre-scale
#pragma unroll
      for (int r = 0; r < 8; ++r) {
        const int m = m0 + mhi + r;
        qs[m * LDA + n] = (__bf16)((acc[r] + bv) * SCALE);
      }
    } else {                  // k half: write transposed per head
      const int nn = n - CH, h = nn >> 5, d = nn & 31;
#pragma unroll
      for (int r = 0; r < 8; ++r) {
        const int m = m0 + mhi + r;
        kts[(h * HD + d) * KLD + m] = (__bf16)(acc[r] + bv);
      }
    }
  }

  // ---- V projection: [64x128] @ [128x128] ----
  for (int t = wave; t < 32; t += 8) {
    const int m0 = (t >> 3) * 16;
    const int n0 = (t & 7) * 16;
    v8f acc = {};
#pragma unroll
    for (int kk = 0; kk < CH; kk += 32) {
      v16bf a = load_a_bf16(xb, m0, LDA, kk, lane);
      v16bf b = load_b_bf16(v_wt, kk, CH, n0, lane);
      acc = __builtin_amdgcn_wmma_f32_16x16x32_bf16(false, a, false, b,
                                                    (short)0, acc, false, false);
    }
    const int n  = n0 + nlo;
    const float bv = v_b[n];
#pragma unroll
    for (int r = 0; r < 8; ++r) {
      const int m = m0 + mhi + r;
      vs[m * LDA + n] = (m < NW) ? (__bf16)(acc[r] + bv) : (__bf16)0.f;
    }
  }
  __syncthreads();

  // ---- attention scores: per head, q(64x32) @ kT(32x64), K=32 = one WMMA ----
  for (int t = wave; t < 64; t += 8) {
    const int h  = t >> 4;
    const int m0 = ((t >> 2) & 3) * 16;
    const int n0 = (t & 3) * 16;
    v8f acc = {};
    v16bf a = load_a_bf16(qs, m0, LDA, h * HD, lane);
    v16bf b = load_b_bf16(kts + h * HD * KLD, 0, KLD, n0, lane);
    acc = __builtin_amdgcn_wmma_f32_16x16x32_bf16(false, a, false, b,
                                                  (short)0, acc, false, false);
    const int n = n0 + nlo;
#pragma unroll
    for (int r = 0; r < 8; ++r) {
      const int m  = m0 + mhi + r;
      const float bv = (m < NW && n < NW) ? bias_full[(h * NW + m) * NW + n] : 0.f;
      sc[(h * NP + m) * NPS + n] = acc[r] + bv;
    }
  }
  __syncthreads();

  // ---- softmax over 49 valid columns, one thread per (head,row) ----
  for (int row = tid; row < NH * NW; row += 256) {
    const int h = row / NW, m = row % NW;
    float* p = sc + (h * NP + m) * NPS;
    float mx = -1e30f;
    for (int j = 0; j < NW; ++j) mx = fmaxf(mx, p[j]);
    float s = 0.f;
    for (int j = 0; j < NW; ++j) { const float e = __expf(p[j] - mx); p[j] = e; s += e; }
    const float inv = 1.f / s;
    for (int j = 0; j < NW; ++j) p[j] *= inv;
    for (int j = NW; j < NP; ++j) p[j] = 0.f;   // kill padded columns
  }
  __syncthreads();

  // ---- P @ V: per head, P(64x64) @ v(64x32); A built from f32 probs ----
  for (int t = wave; t < 32; t += 8) {
    const int h  = t >> 3;
    const int m0 = ((t >> 1) & 3) * 16;
    const int d0 = (t & 1) * 16;
    v8f acc = {};
#pragma unroll
    for (int kk = 0; kk < NP; kk += 32) {
      v16bf a = load_a_f32(sc + h * NP * NPS, m0, NPS, kk, lane);
      v16bf b = load_b_bf16(vs, kk, LDA, h * HD + d0, lane);
      acc = __builtin_amdgcn_wmma_f32_16x16x32_bf16(false, a, false, b,
                                                    (short)0, acc, false, false);
    }
    const int n = h * HD + d0 + nlo;
#pragma unroll
    for (int r = 0; r < 8; ++r) {
      const int m = m0 + mhi + r;
      xb[m * LDA + n] = (__bf16)acc[r];   // xb reused as head-concat buffer
    }
  }
  __syncthreads();

  // ---- output projection: [64x128] @ [128x128] -> global f32 (non-temporal) ----
  float* og = out + (size_t)win * NW * CH;
  for (int t = wave; t < 32; t += 8) {
    const int m0 = (t >> 3) * 16;
    const int n0 = (t & 7) * 16;
    v8f acc = {};
#pragma unroll
    for (int kk = 0; kk < CH; kk += 32) {
      v16bf a = load_a_bf16(xb, m0, LDA, kk, lane);
      v16bf b = load_b_bf16(proj_wt, kk, CH, n0, lane);
      acc = __builtin_amdgcn_wmma_f32_16x16x32_bf16(false, a, false, b,
                                                    (short)0, acc, false, false);
    }
    const int n  = n0 + nlo;
    const float pb = proj_b[n];
#pragma unroll
    for (int r = 0; r < 8; ++r) {
      const int m = m0 + mhi + r;
      if (m < NW) __builtin_nontemporal_store(acc[r] + pb, &og[m * CH + n]);
    }
  }
}

// ---------------- launch ----------------

extern "C" void kernel_launch(void* const* d_in, const int* in_sizes, int n_in,
                              void* d_out, int out_size, void* d_ws, size_t ws_size,
                              hipStream_t stream) {
  (void)in_sizes; (void)n_in; (void)out_size; (void)ws_size;

  const float* x          = (const float*)d_in[0];
  const float* qk_w       = (const float*)d_in[1];
  const float* qk_b       = (const float*)d_in[2];
  const float* v_w        = (const float*)d_in[3];
  const float* v_b        = (const float*)d_in[4];
  const float* proj_w     = (const float*)d_in[5];
  const float* proj_b     = (const float*)d_in[6];
  const float* bias_table = (const float*)d_in[7];
  const int*   rel_idx    = (const int*)d_in[8];
  float* out = (float*)d_out;

  // workspace layout
  char* ws = (char*)d_ws;
  __bf16* qk_wt   = (__bf16*)(ws + 0);            // 128*256*2 = 65536
  __bf16* v_wt    = (__bf16*)(ws + 65536);        // 128*128*2 = 32768
  __bf16* proj_wt = (__bf16*)(ws + 98304);        // 32768
  float*  bias_f  = (float*) (ws + 131072);       // 4*49*49*4 = 38416

  prep_kernel<<<dim3((CH * QKD + 255) / 256), dim3(256), 0, stream>>>(
      qk_w, v_w, proj_w, bias_table, rel_idx, qk_wt, v_wt, proj_wt, bias_f);

  attn_fused_kernel<<<dim3(NWIN), dim3(256), 0, stream>>>(
      x, qk_b, v_b, proj_b, qk_wt, v_wt, proj_wt, bias_f, out);
}
